// MultiHeadAttention_2929167696266
// MI455X (gfx1250) — matrix operand order repaired and checked by reading, not compiled, not run on hardware
//
#include <hip/hip_runtime.h>
#include <hip/hip_bf16.h>
#include <stdint.h>

// MI455X (gfx1250) multi-head attention:
//   - f16 WMMA (v_wmma_f32_16x16x32_f16) for all matmuls, f32 accumulate
//   - projection writes head-major f16 Q/K/V^T so attention tiles are raw
//     byte copies -> staged with GLOBAL_LOAD_ASYNC_TO_LDS_B128 (ASYNCcnt),
//     double-buffered to overlap DMA with WMMA/softmax
//   - exact-f32 online softmax with NEG sentinel, causal block skipping
//   - projection kernel templated on output (Q/K/V) => branch-free epilogue

typedef __attribute__((ext_vector_type(16))) _Float16 v16h;
typedef __attribute__((ext_vector_type(4)))  _Float16 v4h;
typedef __attribute__((ext_vector_type(8)))  float    v8f;

#define HIDDEN 256
#define NHEADS 8
#define HDIM   32
#define SEQ    1024
#define BATCH  8
#define NEGVAL (-4294967295.0f)   // -(2^32)+1 sentinel from reference

// ---------------------------------------------------------------------------
// CDNA5 async global->LDS copy (16B per lane) + async counter wait.
// LDS operand is the byte offset within the workgroup's LDS allocation
// (low 32 bits of the flat shared pointer).
// ---------------------------------------------------------------------------
__device__ __forceinline__ uint32_t lds_off(const void* p) {
  return (uint32_t)(uintptr_t)p;
}
__device__ __forceinline__ void async_copy_b128(uint32_t lds_byte, const void* gptr) {
  asm volatile("global_load_async_to_lds_b128 %0, %1, off"
               :: "v"(lds_byte), "v"((uint64_t)(uintptr_t)gptr)
               : "memory");
}

// ---------------------------------------------------------------------------
// WMMA fragment gathers (CDNA5 ISA VGPR layout tables, wave32).
// A 16x32 f16: lane<16 -> row=lane, K=0..7 (halves 0-7) and K=16..23
// (halves 8-15); lane>=16 -> row=lane-16, K=8..15 and K=24..31.
// ---------------------------------------------------------------------------
__device__ __forceinline__ v16h load_a_frag(const _Float16* A, int ld,
                                            int mBase, int kBase, int lane) {
  const _Float16* row = A + (mBase + (lane & 15)) * ld + kBase + ((lane >> 4) ? 8 : 0);
  v16h a;
#pragma unroll
  for (int i = 0; i < 8; ++i) a[i] = row[i];
#pragma unroll
  for (int i = 0; i < 8; ++i) a[8 + i] = row[16 + i];
  return a;
}
// B 32x16 f16 supplied as B^T (row-major [N][K]): same K order as A fragment.
__device__ __forceinline__ v16h load_bt_frag(const _Float16* Bt, int ld,
                                             int nBase, int kBase, int lane) {
  const _Float16* row = Bt + (nBase + (lane & 15)) * ld + kBase + ((lane >> 4) ? 8 : 0);
  v16h b;
#pragma unroll
  for (int i = 0; i < 8; ++i) b[i] = row[i];
#pragma unroll
  for (int i = 0; i < 8; ++i) b[8 + i] = row[16 + i];
  return b;
}

// ---------------------------------------------------------------------------
// Kernel 1: exact f32 row-sum masks (mask = sign(|sum(row)|))
// ---------------------------------------------------------------------------
__global__ void __launch_bounds__(256)
rowmask_kernel(const float* __restrict__ queries, const float* __restrict__ keys,
               float* __restrict__ qmask, float* __restrict__ kmask) {
  const float* src = (blockIdx.y == 0) ? queries : keys;
  float* dst       = (blockIdx.y == 0) ? qmask   : kmask;
  const int row = blockIdx.x;
  __shared__ float red[256];
  red[threadIdx.x] = src[(size_t)row * HIDDEN + threadIdx.x];
  __syncthreads();
  for (int s = 128; s > 0; s >>= 1) {
    if (threadIdx.x < s) red[threadIdx.x] += red[threadIdx.x + s];
    __syncthreads();
  }
  if (threadIdx.x == 0) dst[row] = (red[0] != 0.0f) ? 1.0f : 0.0f;
}

// ---------------------------------------------------------------------------
// Kernel 2: projection GEMM [8192x256]@[256x256]+bias, WMMA-tiled.
// Templated on WHICH (0=Q, 1=K, 2=V) => branch-free epilogue scatter:
//   WHICH=0: Qh[n][h][t][d] * (1/16)   WHICH=1: Kh[n][h][t][d]
//   WHICH=2: VhT[n][h][d][t]
// grid = (M/64, N/64); 8 waves x two 16x16 tiles; K-loop 8x32.
// ---------------------------------------------------------------------------
template <int WHICH>
__global__ void __launch_bounds__(256)
proj_gemm_kernel(const float* __restrict__ src, const float* __restrict__ W,
                 const float* __restrict__ bias, _Float16* __restrict__ dst) {
  __shared__ alignas(16) _Float16 As[64 * 32];   // [m][k]
  __shared__ alignas(16) _Float16 Bt[64 * 32];   // W^T tile [n][k]

  const int tid  = threadIdx.x;
  const int lane = tid & 31;
  const int wave = tid >> 5;
  const int wm = wave & 3;
  const int wg = wave >> 2;
  const int rowBlock = blockIdx.x * 64;
  const int colBlock = blockIdx.y * 64;

  v8f c0 = {};
  v8f c1 = {};

  for (int kk = 0; kk < 8; ++kk) {
    __syncthreads();
    // A tile: float4 loads (16B aligned), pack 4 halves -> one b64 LDS store
#pragma unroll
    for (int i = 0; i < 2; ++i) {
      const int linear = i * 1024 + tid * 4;
      const int r = linear >> 5, c = linear & 31;
      const float4 f =
          *(const float4*)&src[(size_t)(rowBlock + r) * HIDDEN + kk * 32 + c];
      v4h hv = { (_Float16)f.x, (_Float16)f.y, (_Float16)f.z, (_Float16)f.w };
      *(v4h*)&As[linear] = hv;
    }
    // W^T tile: float4 loads over n (coalesced), 4x b16 scatter (transpose)
#pragma unroll
    for (int i = 0; i < 2; ++i) {
      const int linear = i * 1024 + tid * 4;
      const int n0 = linear & 63, k = linear >> 6;
      const float4 f =
          *(const float4*)&W[(size_t)(kk * 32 + k) * HIDDEN + colBlock + n0];
      Bt[(n0 + 0) * 32 + k] = (_Float16)f.x;
      Bt[(n0 + 1) * 32 + k] = (_Float16)f.y;
      Bt[(n0 + 2) * 32 + k] = (_Float16)f.z;
      Bt[(n0 + 3) * 32 + k] = (_Float16)f.w;
    }
    __syncthreads();

    v16h a  = load_a_frag(As, 32, wm * 16, 0, lane);
    v16h b0 = load_bt_frag(Bt, 32, (wg * 2 + 0) * 16, 0, lane);
    v16h b1 = load_bt_frag(Bt, 32, (wg * 2 + 1) * 16, 0, lane);
    c0 = __builtin_amdgcn_wmma_f32_16x16x32_f16(false, a, false, b0, (short)0, c0, false, false);
    c1 = __builtin_amdgcn_wmma_f32_16x16x32_f16(false, a, false, b1, (short)0, c1, false, false);
  }

  const int hi   = (lane >> 4) ? 8 : 0;
  const int nloc = lane & 15;
#pragma unroll
  for (int t = 0; t < 2; ++t) {
    const int ng = colBlock + (wg * 2 + t) * 16 + nloc;
    const float bb = bias[ng];
    const int hh = ng >> 5;       // head
    const int jj = ng & 31;       // dim within head
    const v8f& c = t ? c1 : c0;
#pragma unroll
    for (int r = 0; r < 8; ++r) {
      const int rg = rowBlock + wm * 16 + r + hi;    // n*SEQ + t
      const int nn = rg >> 10, tt = rg & 1023;
      const size_t nhI = (size_t)(nn * NHEADS + hh);
      const float val = c[r] + bb;
      if (WHICH == 0)
        dst[(nhI * SEQ + tt) * HDIM + jj] = (_Float16)(val * 0.0625f); // fold 1/sqrt(C)
      else if (WHICH == 1)
        dst[(nhI * SEQ + tt) * HDIM + jj] = (_Float16)val;
      else
        dst[(nhI * HDIM + jj) * SEQ + tt] = (_Float16)val;
    }
  }
}

// ---------------------------------------------------------------------------
// Kernel 3: flash attention per (head, batch, 64-query block).
// grid = (SEQ/64, BATCH, NHEADS); 256 threads (8 waves).
// K/V tiles double-buffered via async global->LDS DMA; causal block skip.
// ---------------------------------------------------------------------------
__global__ void __launch_bounds__(256)
attention_kernel(const _Float16* __restrict__ Qh, const _Float16* __restrict__ Kh,
                 const _Float16* __restrict__ VhT,
                 const float* __restrict__ qmask, const float* __restrict__ kmask,
                 const float* __restrict__ queries, float* __restrict__ out) {
  const int qb = blockIdx.x;
  const int n  = blockIdx.y;
  const int h  = blockIdx.z;
  const int tid  = threadIdx.x;
  const int lane = tid & 31;
  const int wave = tid >> 5;

  __shared__ alignas(16) _Float16 Qs[64 * 32];      // [q][d] (pre-scaled f16)
  __shared__ alignas(16) _Float16 Ks[2][64 * 32];   // [key][d]
  __shared__ alignas(16) _Float16 VsT[2][32 * 64];  // [d][key]
  __shared__ alignas(16) _Float16 Ps[64 * 64];      // probabilities f16
  __shared__ float Ss[64 * 64];                     // raw scores f32
  __shared__ float Os[64 * 32];                     // output accumulator f32
  __shared__ float ms[64], ls[64], kms[2][64];
  __shared__ float redmax[64 * 4], redsum[64 * 4];

  const int qBase = qb * 64;
  const size_t nhI = (size_t)(n * NHEADS + h);
  const char* qbase = (const char*)(Qh  + nhI * SEQ * HDIM);
  const char* kbase = (const char*)(Kh  + nhI * SEQ * HDIM);
  const char* vbase = (const char*)(VhT + nhI * HDIM * SEQ);
  const size_t batchOff = (size_t)n * SEQ;

  // ---- prologue: async DMA Q tile + K/V block 0; init state -------------
  async_copy_b128(lds_off(Qs) + (uint32_t)(tid * 16),
                  qbase + (size_t)qBase * (HDIM * 2) + tid * 16);
  async_copy_b128(lds_off(Ks[0]) + (uint32_t)(tid * 16), kbase + tid * 16);
  async_copy_b128(lds_off(VsT[0]) + (uint32_t)((tid >> 3) * 128 + (tid & 7) * 16),
                  vbase + (size_t)(tid >> 3) * (SEQ * 2) + (size_t)((tid & 7) * 8) * 2);
  if (tid < 64) kms[0][tid] = kmask[batchOff + tid];
#pragma unroll
  for (int i = 0; i < 8; ++i) Os[tid + i * 256] = 0.0f;
  if (tid < 64) { ms[tid] = NEGVAL; ls[tid] = 0.0f; }

  const int nKb = qb + 1;  // causal limit
  for (int kb = 0; kb < nKb; ++kb) {
    const int cur = kb & 1;
    const int nxt = cur ^ 1;
    const int kBase = kb * 64;
    const bool hasNext = (kb + 1 < nKb);

    float kmNext = 0.0f;
    if (hasNext) {
      const int kN = kBase + 64;
      async_copy_b128(lds_off(Ks[nxt]) + (uint32_t)(tid * 16),
                      kbase + (size_t)kN * (HDIM * 2) + tid * 16);
      async_copy_b128(lds_off(VsT[nxt]) + (uint32_t)((tid >> 3) * 128 + (tid & 7) * 16),
                      vbase + (size_t)(tid >> 3) * (SEQ * 2) + (size_t)(kN + (tid & 7) * 8) * 2);
      if (tid < 64) kmNext = kmask[batchOff + kN + tid];
      asm volatile("s_wait_asynccnt 0x2" ::: "memory"); // leave only 2 newest in flight
    } else {
      asm volatile("s_wait_asynccnt 0x0" ::: "memory");
    }
    __syncthreads();  // B1: current tiles resident for whole workgroup

    // ---- S = Qs @ Ks^T : 4x4 tiles of 16x16, two per wave ---------------
    {
      const int tm = wave & 3;
      const int tg = wave >> 2;
      v16h a = load_a_frag(Qs, 32, tm * 16, 0, lane);
      const int rowOff = tm * 16 + ((lane >> 4) ? 8 : 0);
#pragma unroll
      for (int t = 0; t < 2; ++t) {
        const int tn = tg * 2 + t;
        v16h b = load_bt_frag(Ks[cur], 32, tn * 16, 0, lane);
        v8f c = {};
        c = __builtin_amdgcn_wmma_f32_16x16x32_f16(false, a, false, b, (short)0, c, false, false);
        const int col = tn * 16 + (lane & 15);
#pragma unroll
        for (int r = 0; r < 8; ++r) Ss[(rowOff + r) * 64 + col] = c[r];
      }
    }
    __syncthreads();  // B2

    // ---- online softmax: 4 threads per row, 16 cols each ----------------
    const int r  = tid >> 2;
    const int qp = tid & 3;
    const int c0 = qp * 16;
    float sreg[16];
    const float mOld = ms[r];
    float localmax = NEGVAL;
#pragma unroll
    for (int i = 0; i < 16; ++i) {
      float s = Ss[r * 64 + c0 + i];
      if ((kBase + c0 + i) > (qBase + r) || kms[cur][c0 + i] == 0.0f) s = NEGVAL;
      sreg[i] = s;
      localmax = fmaxf(localmax, s);
    }
    redmax[tid] = localmax;
    __syncthreads();  // B3

    const float mNew = fmaxf(mOld,
        fmaxf(fmaxf(redmax[r * 4 + 0], redmax[r * 4 + 1]),
              fmaxf(redmax[r * 4 + 2], redmax[r * 4 + 3])));
    const float scale = __expf(mOld - mNew);  // first block: exp(NEG-m) -> 0
    float psum = 0.0f;
#pragma unroll
    for (int i = 0; i < 16; ++i) {
      float p = __expf(sreg[i] - mNew);       // masked: underflow -> 0 exactly
      psum += p;
      Ps[r * 64 + c0 + i] = (_Float16)p;
    }
    redsum[tid] = psum;
#pragma unroll
    for (int j = 0; j < 8; ++j) Os[r * 32 + qp * 8 + j] *= scale;  // own slice
    if (hasNext && tid < 64) kms[nxt][tid] = kmNext;
    __syncthreads();  // B4

    if (qp == 0) {
      ls[r] = ls[r] * scale +
              (redsum[r * 4 + 0] + redsum[r * 4 + 1] +
               redsum[r * 4 + 2] + redsum[r * 4 + 3]);
      ms[r] = mNew;
    }

    // ---- O += P @ V : 4x2 tiles, one per wave; K=64 -> 2 WMMA steps -----
    {
      const int tm = wave & 3;
      const int tn = wave >> 2;
      const int rowOff = tm * 16 + ((lane >> 4) ? 8 : 0);
      const int col = tn * 16 + (lane & 15);
      v8f c;
#pragma unroll
      for (int rr = 0; rr < 8; ++rr) c[rr] = Os[(rowOff + rr) * 32 + col];
#pragma unroll
      for (int ks = 0; ks < 2; ++ks) {
        v16h a = load_a_frag(Ps, 64, tm * 16, ks * 32, lane);
        v16h b = load_bt_frag(VsT[cur], 64, tn * 16, ks * 32, lane);
        c = __builtin_amdgcn_wmma_f32_16x16x32_f16(false, a, false, b, (short)0, c, false, false);
      }
#pragma unroll
      for (int rr = 0; rr < 8; ++rr) Os[(rowOff + rr) * 32 + col] = c[rr];
    }
  }
  __syncthreads();

  // ---- epilogue: normalize, post-softmax query mask, residual ----------
#pragma unroll
  for (int i = 0; i < 8; ++i) {
    const int idx = tid + i * 256;
    const int rr = idx >> 5, jj = idx & 31;
    const float l   = ls[rr];
    const float inv = (l > 0.0f) ? (1.0f / l) : 0.0f;
    const float qm  = qmask[batchOff + qBase + rr];
    const size_t base = (batchOff + qBase + rr) * (size_t)HIDDEN + (size_t)h * HDIM;
    out[base + jj] = Os[rr * 32 + jj] * inv * qm + queries[base + jj];
  }
}

// ---------------------------------------------------------------------------
extern "C" void kernel_launch(void* const* d_in, const int* in_sizes, int n_in,
                              void* d_out, int out_size, void* d_ws, size_t ws_size,
                              hipStream_t stream) {
  const float* queries = (const float*)d_in[0];
  const float* keys    = (const float*)d_in[1];
  const float* Wq = (const float*)d_in[2];
  const float* bq = (const float*)d_in[3];
  const float* Wk = (const float*)d_in[4];
  const float* bk = (const float*)d_in[5];
  const float* Wv = (const float*)d_in[6];
  const float* bv = (const float*)d_in[7];
  float* out = (float*)d_out;

  // Workspace: Qh | Kh | VhT (f16, 4 MB each) | qmask | kmask (~12.6 MB total)
  const size_t elems = (size_t)BATCH * SEQ * HIDDEN;
  _Float16* Qh  = (_Float16*)d_ws;
  _Float16* Kh  = Qh + elems;
  _Float16* VhT = Kh + elems;
  float* qm = (float*)(VhT + elems);
  float* km = qm + (size_t)BATCH * SEQ;

  const dim3 gemmGrid((BATCH * SEQ) / 64, HIDDEN / 64);
  rowmask_kernel<<<dim3(BATCH * SEQ, 2), 256, 0, stream>>>(queries, keys, qm, km);
  proj_gemm_kernel<0><<<gemmGrid, 256, 0, stream>>>(queries, Wq, bq, Qh);
  proj_gemm_kernel<1><<<gemmGrid, 256, 0, stream>>>(keys,    Wk, bk, Kh);
  proj_gemm_kernel<2><<<gemmGrid, 256, 0, stream>>>(keys,    Wv, bv, VhT);
  attention_kernel<<<dim3(SEQ / 64, BATCH, NHEADS), 256, 0, stream>>>(
      Qh, Kh, VhT, qm, km, queries, out);
}
